// Attention_77876347011235
// MI455X (gfx1250) — matrix operand order repaired and checked by reading, not compiled, not run on hardware
//
#include <hip/hip_runtime.h>
#include <math.h>

// ---------------------------------------------------------------------------
// Swin-style window attention with residual scores, MI455X (gfx1250, wave32).
// All matrix math goes through v_wmma_f32_16x16x32_f16 (f16 operands, f32
// accumulation). Per-window data (x, Q, K, V^T, P) lives entirely in LDS.
// A-panels are 64 rows (rows 49..63 zero-filled once), so every WMMA
// A-fragment load is two unconditional ds_load_b128 -- no exec masking in
// the GEMM inner loops.
// ---------------------------------------------------------------------------

typedef __attribute__((ext_vector_type(16))) _Float16 v16h;
typedef __attribute__((ext_vector_type(8)))  _Float16 v8h;
typedef __attribute__((ext_vector_type(4)))  _Float16 v4h;
typedef __attribute__((ext_vector_type(8)))  float    v8f;
typedef __attribute__((ext_vector_type(4)))  float    v4f;
typedef __attribute__((ext_vector_type(4)))  unsigned int v4u;

#define NTOK 49          // tokens per window (7x7)
#define CDIM 768
#define HQ   12          // heads
#define HD   64          // head dim
#define KC   192         // K-chunk staged in LDS (4 chunks of 768)
#define XS_STRIDE 200    // halves: KC + 8 pad (16B row align, 36-dword stagger)
#define QK_STRIDE 72     // halves: 64 + 8 pad (16B row align, conflict-free)
#define P_STRIDE  72

__device__ __forceinline__ v8f wmma16(v16h a, v16h b, v8f c) {
  // D = A(16x32 f16) * B(32x16 f16) + C(16x16 f32)
  return __builtin_amdgcn_wmma_f32_16x16x32_f16(false, a, false, b,
                                                (short)0, c, false, false);
}

__device__ __forceinline__ v8f zero_c() {
  v8f z;
#pragma unroll
  for (int i = 0; i < 8; ++i) z[i] = 0.0f;
  return z;
}

// A fragment (16x32, f16) from row-major f16 LDS. Contiguous K layout:
// lanes 0-15: M=lane,    halves 0..15 = K k0+0..15
// lanes16-31: M=lane-16, halves 0..15 = K k0+16..31
__device__ __forceinline__ v16h load_a(const _Float16* base, int stride,
                                       int m0, int k0) {
  const int lane = threadIdx.x & 31;
  const int m  = m0 + (lane & 15);
  const int kb = k0 + ((lane >> 4) << 4);
  v8h lo = *(const v8h*)(base + m * stride + kb);
  v8h hi = *(const v8h*)(base + m * stride + kb + 8);
  v16h a;
#pragma unroll
  for (int i = 0; i < 8; ++i) { a[i] = lo[i]; a[8 + i] = hi[i]; }
  return a;
}

// B fragment (32x16, f16) from K-contiguous ("transposed") storage T[n][k].
// Contiguous K layout (same as load_a): lanes 0-15: N=lane, halves 0..15 = K k0+0..15;
//             lanes16-31: N=lane-16, halves = K k0+16..31.
__device__ __forceinline__ v16h load_bT(const _Float16* base, int stride,
                                        int n0, int k0) {
  const int lane = threadIdx.x & 31;
  const int n  = n0 + (lane & 15);
  const int kb = k0 + ((lane >> 4) << 4);
  return *(const v16h*)(base + (size_t)n * stride + kb);
}

// ---------------------------------------------------------------------------
// Weight prep: f32 [k][n] -> f16 [n][k] (K-contiguous for B fragments).
// ---------------------------------------------------------------------------
__global__ void prep_weights(const float* __restrict__ qkv_w,
                             const float* __restrict__ proj_w,
                             _Float16* __restrict__ qkv_wT,
                             _Float16* __restrict__ proj_wT) {
  const int n1 = 3 * CDIM * CDIM;   // 2304*768
  const int n2 = CDIM * CDIM;
  for (int i = blockIdx.x * blockDim.x + threadIdx.x; i < n1 + n2;
       i += gridDim.x * blockDim.x) {
    if (i < n1) {
      const int nrow = i / CDIM, k = i - nrow * CDIM;
      qkv_wT[i] = (_Float16)qkv_w[(size_t)k * (3 * CDIM) + nrow];
    } else {
      const int j = i - n1;
      const int nrow = j / CDIM, k = j - nrow * CDIM;
      proj_wT[j] = (_Float16)proj_w[(size_t)k * CDIM + nrow];
    }
  }
}

// Bias pre-permuted into WMMA C/D lane order:
// bias_perm[h][strip][nt][lane][r] -> one v8f load per 16x16 tile.
__global__ void prep_bias(const float* __restrict__ bias_table,
                          const int* __restrict__ rel_idx,
                          float* __restrict__ bias_perm) {
  const int h = blockIdx.x;
  for (int i = threadIdx.x; i < 4096; i += blockDim.x) {
    const int r    = i & 7;
    const int lane = (i >> 3) & 31;
    const int nt   = (i >> 8) & 3;
    const int s    = i >> 10;
    const int m = s * 16 + ((lane >> 4) << 3) + r;
    const int n = nt * 16 + (lane & 15);
    float v = 0.0f;
    if (m < NTOK && n < NTOK) v = bias_table[rel_idx[m * NTOK + n] * HQ + h];
    bias_perm[(size_t)h * 4096 + i] = v;
  }
}

// ---------------------------------------------------------------------------
// Fused per-window kernel: qkv GEMM + bias/prev + softmax + P@V.
// One workgroup (8 waves) per window b. Static LDS = 62,464 B (< 64 KB).
// ---------------------------------------------------------------------------
__global__ __launch_bounds__(256)
void attn_fused(const float* __restrict__ x, const float* __restrict__ prev,
                const _Float16* __restrict__ qkv_wT,
                const float* __restrict__ bias_perm,
                _Float16* __restrict__ out_ws, float* __restrict__ prev_out) {
  __shared__ __align__(32) _Float16 xs [64 * XS_STRIDE];       // 25,600 B
  __shared__ __align__(32) _Float16 qs [64 * QK_STRIDE];       //  9,216 B
  __shared__ __align__(32) _Float16 ksm[64 * QK_STRIDE];       //  9,216 B
  __shared__ __align__(32) _Float16 vts[64 * QK_STRIDE];       //  9,216 B (V^T)
  __shared__ __align__(32) _Float16 pbuf[4 * 16 * P_STRIDE];   //  9,216 B

  const int b    = blockIdx.x;
  const int tid  = threadIdx.x;
  const int wave = tid >> 5;
  const int lane = tid & 31;
  const int nl   = lane & 15;
  const int moff = (lane >> 4) << 3;

  const float* xg    = x + (size_t)b * NTOK * CDIM;
  _Float16*    outB  = out_ws + (size_t)b * NTOK * CDIM;

  // Zero rows 49..63 of the A-panel once; staging below only rewrites 0..48,
  // so GEMM A-fragments never need a bounds guard.
  {
    v4h z;
#pragma unroll
    for (int j = 0; j < 4; ++j) z[j] = (_Float16)0.0f;
    for (int i = tid; i < 15 * (XS_STRIDE / 4); i += 256) {
      const int r = 49 + i / (XS_STRIDE / 4), c = i % (XS_STRIDE / 4);
      *(v4h*)(xs + r * XS_STRIDE + c * 4) = z;
    }
  }

  for (int h = 0; h < HQ; ++h) {
    // ---------------- QKV GEMM for head h --------------------------------
    // 12 N-tiles (q:0-3, k:4-7, v:8-11); wave w owns nt=w and (w<4) nt=w+8.
    v8f acc[2][4];
#pragma unroll
    for (int q = 0; q < 2; ++q)
#pragma unroll
      for (int mt = 0; mt < 4; ++mt) acc[q][mt] = zero_c();

    for (int kc = 0; kc < CDIM / KC; ++kc) {
      // stage x[b][:, kc*192 .. +192) as f16 into LDS (float4 -> packed f16)
      for (int i = tid; i < NTOK * (KC / 4); i += 256) {
        const int r = i / (KC / 4), c = i % (KC / 4);   // c in float4 units
        v4f f = *(const v4f*)(xg + (size_t)r * CDIM + kc * KC + c * 4);
        v4h hh;
#pragma unroll
        for (int j = 0; j < 4; ++j) hh[j] = (_Float16)f[j];
        *(v4h*)(xs + r * XS_STRIDE + c * 4) = hh;
      }
      __syncthreads();
#pragma unroll
      for (int q = 0; q < 2; ++q) {
        if (q == 0 || wave < 4) {
          const int nt      = wave + q * 8;
          const int sec     = nt >> 2;                        // 0=q,1=k,2=v
          const int colBase = sec * CDIM + h * HD + (nt & 3) * 16;
#pragma unroll 2
          for (int ks = 0; ks < KC; ks += 32) {
            v16h bf = load_bT(qkv_wT, CDIM, colBase, kc * KC + ks);
#pragma unroll
            for (int mt = 0; mt < 4; ++mt) {
              v16h af = load_a(xs, XS_STRIDE, mt * 16, ks);
              acc[q][mt] = wmma16(af, bf, acc[q][mt]);
            }
          }
        }
      }
      __syncthreads();
    }
    // scatter results into qs / ksm / vts (f16; rows >=49 are exact zeros
    // because the xs pad rows are zero). q pre-scaled by 1/sqrt(64)=0.125.
#pragma unroll
    for (int q = 0; q < 2; ++q) {
      if (q == 0 || wave < 4) {
        const int nt  = wave + q * 8;
        const int sec = nt >> 2;
        const int nc  = (nt & 3) * 16 + nl;
#pragma unroll
        for (int mt = 0; mt < 4; ++mt)
#pragma unroll
          for (int r = 0; r < 8; ++r) {
            const int m = mt * 16 + moff + r;
            const float vv = acc[q][mt][r];
            if (sec == 0)      qs [m * QK_STRIDE + nc] = (_Float16)(vv * 0.125f);
            else if (sec == 1) ksm[m * QK_STRIDE + nc] = (_Float16)vv;
            else               vts[nc * QK_STRIDE + m] = (_Float16)vv;  // V^T
          }
      }
    }
    __syncthreads();

    // ---------------- attention: waves 0..3, 16-row strips ----------------
    if (wave < 4) {
      const int m0 = wave * 16;
      v16h qa0 = load_a(qs, QK_STRIDE, m0, 0);
      v16h qa1 = load_a(qs, QK_STRIDE, m0, 32);
      const float* bstrip    = bias_perm + (((size_t)h * 4 + wave) * 4) * 256;
      const float* prevBH    = prev     + ((size_t)b * HQ + h) * NTOK * NTOK;
      float*       prevOutBH = prev_out + ((size_t)b * HQ + h) * NTOK * NTOK;

      v8f sc[4];
#pragma unroll
      for (int nt = 0; nt < 4; ++nt) {
        v8f c = zero_c();
        c = wmma16(qa0, load_bT(ksm, QK_STRIDE, nt * 16, 0),  c);  // S=Q*K^T
        c = wmma16(qa1, load_bT(ksm, QK_STRIDE, nt * 16, 32), c);
        v8f bv = *(const v8f*)(bstrip + nt * 256 + lane * 8);      // bias tile
        const int n = nt * 16 + nl;
        if (wave < 3) {                       // all 16 rows valid (uniform)
          if (nt < 3) {                       // all 16 cols valid
#pragma unroll
            for (int r = 0; r < 8; ++r) {
              const int m = m0 + moff + r;
              float s = c[r] + bv[r] + prevBH[m * NTOK + n];
              prevOutBH[m * NTOK + n] = s;
              c[r] = s;
            }
          } else {                            // only col 48 (nl==0) valid
#pragma unroll
            for (int r = 0; r < 8; ++r) {
              const int m = m0 + moff + r;
              float s = c[r] + bv[r];
              if (nl == 0) { s += prevBH[m * NTOK + 48]; prevOutBH[m * NTOK + 48] = s; }
              c[r] = (nl == 0) ? s : -3.0e38f;
            }
          }
        } else {                              // wave 3: only row 48 valid
#pragma unroll
          for (int r = 0; r < 8; ++r) {
            float s = c[r] + bv[r];
            const bool mok = (moff + r == 0);
            const bool nok = (nt < 3) || (nl == 0);
            if (mok && nok) { s += prevBH[48 * NTOK + n]; prevOutBH[48 * NTOK + n] = s; }
            c[r] = nok ? s : -3.0e38f;
          }
        }
        sc[nt] = c;
      }

      // In-register softmax per row: D layout => fixed column per lane, so a
      // row spans the 16 lanes of a half-wave (width-16 shfl_xor reductions).
      _Float16* prow = pbuf + wave * 16 * P_STRIDE;
#pragma unroll
      for (int r = 0; r < 8; ++r) {
        float mx = fmaxf(fmaxf(sc[0][r], sc[1][r]), fmaxf(sc[2][r], sc[3][r]));
        mx = fmaxf(mx, __shfl_xor(mx, 1, 16));
        mx = fmaxf(mx, __shfl_xor(mx, 2, 16));
        mx = fmaxf(mx, __shfl_xor(mx, 4, 16));
        mx = fmaxf(mx, __shfl_xor(mx, 8, 16));
        const float e0 = __expf(sc[0][r] - mx);
        const float e1 = __expf(sc[1][r] - mx);
        const float e2 = __expf(sc[2][r] - mx);
        const float e3 = __expf(sc[3][r] - mx);
        float sum = e0 + e1 + e2 + e3;
        sum += __shfl_xor(sum, 1, 16);
        sum += __shfl_xor(sum, 2, 16);
        sum += __shfl_xor(sum, 4, 16);
        sum += __shfl_xor(sum, 8, 16);
        const float inv = 1.0f / sum;                    // masked lanes -> 0
        const int mr = (moff + r) * P_STRIDE;
        prow[mr +  0 + nl] = (_Float16)(e0 * inv);
        prow[mr + 16 + nl] = (_Float16)(e1 * inv);
        prow[mr + 32 + nl] = (_Float16)(e2 * inv);
        prow[mr + 48 + nl] = (_Float16)(e3 * inv);
      }

      // O = P @ V ; vts = V^T gives B fragments as contiguous loads.
      v16h pa0 = load_a(prow, P_STRIDE, 0, 0);
      v16h pa1 = load_a(prow, P_STRIDE, 0, 32);
#pragma unroll
      for (int nt = 0; nt < 4; ++nt) {
        v8f c = zero_c();
        c = wmma16(pa0, load_bT(vts, QK_STRIDE, nt * 16, 0),  c);
        c = wmma16(pa1, load_bT(vts, QK_STRIDE, nt * 16, 32), c);
        const int n = nt * 16 + nl;
        if (wave < 3) {
#pragma unroll
          for (int r = 0; r < 8; ++r)
            outB[(size_t)(m0 + moff + r) * CDIM + h * HD + n] = (_Float16)c[r];
        } else if (lane < 16) {               // row 48 only
          outB[(size_t)48 * CDIM + h * HD + n] = (_Float16)c[0];
        }
      }
    }
    __syncthreads();   // qs/ksm/vts reused next head
  }
}

// ---------------------------------------------------------------------------
// Projection: out[b] = O[b](f16) @ proj_wT^T + proj_b, f32 result.
// 3 N-passes x (8 waves x 2 N-tiles x 4 M-tiles), K staged in LDS chunks.
// ---------------------------------------------------------------------------
__global__ __launch_bounds__(256)
void proj_kernel(const _Float16* __restrict__ out_ws,
                 const _Float16* __restrict__ proj_wT,
                 const float* __restrict__ proj_b,
                 float* __restrict__ out) {
  __shared__ __align__(32) _Float16 os[64 * XS_STRIDE];        // 25,600 B
  const int b    = blockIdx.x;
  const int tid  = threadIdx.x;
  const int wave = tid >> 5;
  const int lane = tid & 31;
  const int nl   = lane & 15;
  const int moff = (lane >> 4) << 3;
  const _Float16* ob   = out_ws + (size_t)b * NTOK * CDIM;
  float*          outB = out    + (size_t)b * NTOK * CDIM;

  {  // zero pad rows once (staging below only rewrites rows 0..48)
    v4h z;
#pragma unroll
    for (int j = 0; j < 4; ++j) z[j] = (_Float16)0.0f;
    for (int i = tid; i < 15 * (XS_STRIDE / 4); i += 256) {
      const int r = 49 + i / (XS_STRIDE / 4), c = i % (XS_STRIDE / 4);
      *(v4h*)(os + r * XS_STRIDE + c * 4) = z;
    }
  }

  for (int pass = 0; pass < 3; ++pass) {
    v8f acc[2][4];
#pragma unroll
    for (int q = 0; q < 2; ++q)
#pragma unroll
      for (int mt = 0; mt < 4; ++mt) acc[q][mt] = zero_c();

    for (int kc = 0; kc < CDIM / KC; ++kc) {
      // 16-byte vector copies: 24 uint4 per row of 192 halves
      for (int i = tid; i < NTOK * (KC / 8); i += 256) {
        const int r = i / (KC / 8), c = i % (KC / 8);
        *(v4u*)(os + r * XS_STRIDE + c * 8) =
            *(const v4u*)(ob + (size_t)r * CDIM + kc * KC + c * 8);
      }
      __syncthreads();
#pragma unroll
      for (int q = 0; q < 2; ++q) {
        const int n0 = (pass * 16 + wave + q * 8) * 16;
#pragma unroll 2
        for (int ks = 0; ks < KC; ks += 32) {
          v16h bf = load_bT(proj_wT, CDIM, n0, kc * KC + ks);
#pragma unroll
          for (int mt = 0; mt < 4; ++mt) {
            v16h af = load_a(os, XS_STRIDE, mt * 16, ks);
            acc[q][mt] = wmma16(af, bf, acc[q][mt]);
          }
        }
      }
      __syncthreads();
    }
#pragma unroll
    for (int q = 0; q < 2; ++q) {
      const int n  = (pass * 16 + wave + q * 8) * 16 + nl;
      const float pb = proj_b[n];
#pragma unroll
      for (int mt = 0; mt < 3; ++mt)                    // rows 0..47: no guard
#pragma unroll
        for (int r = 0; r < 8; ++r)
          outB[(size_t)(mt * 16 + moff + r) * CDIM + n] = acc[q][mt][r] + pb;
      if (lane < 16)                                    // row 48 only
        outB[(size_t)48 * CDIM + n] = acc[q][3][0] + pb;
    }
  }
}

// ---------------------------------------------------------------------------
extern "C" void kernel_launch(void* const* d_in, const int* in_sizes, int n_in,
                              void* d_out, int out_size, void* d_ws, size_t ws_size,
                              hipStream_t stream) {
  (void)n_in; (void)out_size; (void)ws_size;
  const float* x          = (const float*)d_in[0];
  const float* prev       = (const float*)d_in[1];
  const float* qkv_w      = (const float*)d_in[2];
  const float* proj_w     = (const float*)d_in[3];
  const float* proj_b     = (const float*)d_in[4];
  const float* bias_table = (const float*)d_in[5];
  const int*   rel_idx    = (const int*)d_in[6];

  const int B = in_sizes[0] / (NTOK * CDIM);                 // 1024

  float* out      = (float*)d_out;
  float* prev_out = out + (size_t)B * NTOK * CDIM;           // tuple order

  // Workspace layout (all offsets 256B-aligned); total ~82 MB.
  char* ws = (char*)d_ws;
  _Float16* qkv_wT   = (_Float16*)(ws);                                  // 3,538,944 B
  _Float16* proj_wT  = (_Float16*)(ws + 3538944);                        // 1,179,648 B
  float*    bias_perm= (float*)   (ws + 3538944 + 1179648);              //   196,608 B
  _Float16* out_ws   = (_Float16*)(ws + 3538944 + 1179648 + 196608);     // B*49*768*2 B

  prep_weights<<<4608, 512, 0, stream>>>(qkv_w, proj_w, qkv_wT, proj_wT);
  prep_bias<<<HQ, 256, 0, stream>>>(bias_table, rel_idx, bias_perm);
  attn_fused<<<B, 256, 0, stream>>>(x, prev, qkv_wT, bias_perm, out_ws, prev_out);
  proj_kernel<<<B, 256, 0, stream>>>(out_ws, proj_wT, proj_b, out);
}
